// SimpleRNN_22376779612691
// MI455X (gfx1250) — compile-verified, matrix operand order checked by reading, not run on hardware
//
#include <hip/hip_runtime.h>

// ---------------------------------------------------------------------------
// SimpleRNN for MI455X (gfx1250, wave32, WMMA)
//   phase 1: xh = X(16384x512) @ Wxh^T(512x1024) + (Wxh_b+Whh_b+bh)   [WMMA bf16]
//   phase 2: h_t = tanh(xh_t + Whh h_{t-1})  16384 sequential steps   [VGPR-pinned weights + grid barrier]
//   phase 3: logits = outs @ fc_w^T + fc_b -> row softmax             [WMMA bf16]
// ---------------------------------------------------------------------------

#define SEQ 16384
#define IN  512
#define HID 1024
#define OUT 256
#define SCAN_NB 32   // 32 blocks x 256 thr; each block owns 32 rows of Whh

typedef __bf16 bf16_t;
typedef __attribute__((ext_vector_type(16))) __bf16 v16bf;
typedef __attribute__((ext_vector_type(8)))  __bf16 v8bf;
typedef __attribute__((ext_vector_type(8)))  float  v8f;

// ---------------- helpers ----------------

__device__ __forceinline__ v16bf mk_frag(const bf16_t* __restrict__ p) {
  // 16-bit 16x32 fragment, per-lane: two contiguous 16B chunks (K..K+7, K+16..K+23)
  v8bf lo = *(const v8bf*)(p);
  v8bf hi = *(const v8bf*)(p + 16);
  v16bf f;
#pragma unroll
  for (int i = 0; i < 8; ++i) { f[i] = lo[i]; f[i + 8] = hi[i]; }
  return f;
}

__device__ __forceinline__ float bflo(unsigned p) { return __uint_as_float(p << 16); }
__device__ __forceinline__ float bfhi(unsigned p) { return __uint_as_float(p & 0xffff0000u); }

// ---------------- f32 -> bf16 convert ----------------

__global__ __launch_bounds__(256) void cvt_f32_bf16(const float* __restrict__ s,
                                                    bf16_t* __restrict__ d, int n) {
  int i = blockIdx.x * blockDim.x + threadIdx.x;
  int stride = gridDim.x * blockDim.x;
  for (; i < n; i += stride) d[i] = (bf16_t)s[i];
}

// ---------------- init: fused bias, h0, barrier counter ----------------

__global__ __launch_bounds__(1024) void rnn_init(const float* __restrict__ wxhb,
                                                 const float* __restrict__ whhb,
                                                 const float* __restrict__ bh,
                                                 const float* __restrict__ h0,
                                                 float* __restrict__ bsum,
                                                 float* __restrict__ hbuf,
                                                 unsigned* __restrict__ cnt) {
  int i = threadIdx.x;
  bsum[i] = wxhb[i] + whhb[i] + bh[i];
  hbuf[i] = h0[i];
  hbuf[HID + i] = 0.0f;
  if (i == 0) *cnt = 0u;
}

// ---------------- WMMA GEMM: C(MxN) = A(MxK) @ Bt(NxK)^T + bias ----------------
// block = 256 thr (8 waves), tile 128(M) x 64(N), wave -> 32x32 (2x2 WMMA accs)

__global__ __launch_bounds__(256) void wmma_gemm_bias(const bf16_t* __restrict__ A,
                                                      const bf16_t* __restrict__ Bt,
                                                      const float* __restrict__ bias,
                                                      float* __restrict__ C,
                                                      int M, int N, int K) {
  const int lane  = threadIdx.x & 31;
  const int wave  = threadIdx.x >> 5;
  const int wm    = wave >> 1;        // 0..3
  const int wn    = wave & 1;         // 0..1
  const int m0    = blockIdx.y * 128 + wm * 32;
  const int n0    = blockIdx.x * 64  + wn * 32;
  const int laneM = lane & 15;
  const int kOff  = (lane >> 4) << 3; // 0 or 8

  const bf16_t* pa = A  + (size_t)(m0 + laneM) * K + kOff;
  const bf16_t* pb = Bt + (size_t)(n0 + laneM) * K + kOff;
  const size_t rowStep = (size_t)16 * K;

  v8f acc00 = {}, acc01 = {}, acc10 = {}, acc11 = {};

  for (int k0 = 0; k0 < K; k0 += 32) {
    if (k0 + 32 < K) {                       // pull next K-tile toward the WGP
      __builtin_prefetch(pa + k0 + 32, 0, 1);
      __builtin_prefetch(pb + k0 + 32, 0, 1);
    }
    v16bf a0 = mk_frag(pa + k0);
    v16bf a1 = mk_frag(pa + rowStep + k0);
    v16bf b0 = mk_frag(pb + k0);
    v16bf b1 = mk_frag(pb + rowStep + k0);
    acc00 = __builtin_amdgcn_wmma_f32_16x16x32_bf16(false, a0, false, b0, (short)0, acc00, false, false);
    acc01 = __builtin_amdgcn_wmma_f32_16x16x32_bf16(false, a0, false, b1, (short)0, acc01, false, false);
    acc10 = __builtin_amdgcn_wmma_f32_16x16x32_bf16(false, a1, false, b0, (short)0, acc10, false, false);
    acc11 = __builtin_amdgcn_wmma_f32_16x16x32_bf16(false, a1, false, b1, (short)0, acc11, false, false);
  }

  // C layout per 16x16 tile: elem e -> row = mbase + e + (lane>=16 ? 8 : 0), col = nbase + (lane&15)
  const int rsel = (lane >> 4) * 8;
  const int colA = n0 + laneM;
  const int colB = colA + 16;
  const float bvA = bias ? bias[colA] : 0.0f;
  const float bvB = bias ? bias[colB] : 0.0f;
#pragma unroll
  for (int e = 0; e < 8; ++e) {
    int r0 = m0 + rsel + e;
    int r1 = m0 + 16 + rsel + e;
    C[(size_t)r0 * N + colA] = acc00[e] + bvA;
    C[(size_t)r0 * N + colB] = acc01[e] + bvB;
    C[(size_t)r1 * N + colA] = acc10[e] + bvA;
    C[(size_t)r1 * N + colB] = acc11[e] + bvB;
  }
}

// ---------------- sequential scan ----------------
// 32 blocks x 256 thr. Thread (row=tid>>3, sub=tid&7) keeps 128 bf16 weights of
// Whh[row, sub*128 .. +128) pinned in 64 packed VGPRs. Per step: 32 float4 L2
// reads of h, 128 FMAs, 8-lane shfl reduce, tanh, then device-wide barrier.

__global__ __launch_bounds__(256) void rnn_scan(const float* __restrict__ xh,
                                                const unsigned* __restrict__ WhhP,
                                                float* __restrict__ hbuf,     // 2*HID doubles-buffered
                                                bf16_t* __restrict__ outs,
                                                unsigned* __restrict__ cnt,
                                                int seq) {
  const int tid = threadIdx.x;
  const int gr  = blockIdx.x * 32 + (tid >> 3);  // global hidden row 0..1023
  const int sub = tid & 7;                       // K-slice 0..7

  unsigned w[64];                                // 128 bf16 weights, packed
  const unsigned* wp = WhhP + ((size_t)gr * HID + sub * 128) / 2;
#pragma unroll
  for (int i = 0; i < 64; ++i) w[i] = wp[i];

  for (int t = 0; t < seq; ++t) {
    const float4* h4p = (const float4*)(hbuf + (t & 1) * HID + sub * 128);
    float acc = 0.0f;
#pragma unroll
    for (int i = 0; i < 32; ++i) {
      float4 h4 = h4p[i];
      unsigned p0 = w[2 * i], p1 = w[2 * i + 1];
      acc = fmaf(bflo(p0), h4.x, acc);
      acc = fmaf(bfhi(p0), h4.y, acc);
      acc = fmaf(bflo(p1), h4.z, acc);
      acc = fmaf(bfhi(p1), h4.w, acc);
    }
    acc += __shfl_xor(acc, 1, 32);
    acc += __shfl_xor(acc, 2, 32);
    acc += __shfl_xor(acc, 4, 32);
    if (sub == 0) {
      float v = tanhf(xh[(size_t)t * HID + gr] + acc);
      hbuf[((t + 1) & 1) * HID + gr] = v;
      outs[(size_t)t * HID + gr] = (bf16_t)v;
    }
    // device-wide barrier (monotonic counter, zeroed each launch by rnn_init)
    __threadfence();            // release this thread's h stores to L2
    __syncthreads();
    if (tid == 0) {
      atomicAdd(cnt, 1u);
      unsigned target = (unsigned)(SCAN_NB) * (unsigned)(t + 1);
      while (__hip_atomic_load(cnt, __ATOMIC_RELAXED, __HIP_MEMORY_SCOPE_AGENT) < target) { }
    }
    __syncthreads();
    __threadfence();            // acquire: don't serve stale h from near caches
  }
}

// ---------------- softmax over 256 columns, one block per row ----------------

__global__ __launch_bounds__(256) void softmax256(const float* __restrict__ L,
                                                  float* __restrict__ O) {
  __shared__ float red[256];
  const int row = blockIdx.x, tid = threadIdx.x;
  float v = L[(size_t)row * OUT + tid];
  red[tid] = v;
  __syncthreads();
  for (int s = 128; s > 0; s >>= 1) { if (tid < s) red[tid] = fmaxf(red[tid], red[tid + s]); __syncthreads(); }
  float m = red[0];
  __syncthreads();
  float e = expf(v - m);
  red[tid] = e;
  __syncthreads();
  for (int s = 128; s > 0; s >>= 1) { if (tid < s) red[tid] += red[tid + s]; __syncthreads(); }
  O[(size_t)row * OUT + tid] = e / red[0];
}

// ---------------- launch ----------------

extern "C" void kernel_launch(void* const* d_in, const int* in_sizes, int n_in,
                              void* d_out, int out_size, void* d_ws, size_t ws_size,
                              hipStream_t stream) {
  const float* X     = (const float*)d_in[0];
  const float* h0    = (const float*)d_in[1];
  const float* Wxh   = (const float*)d_in[2];
  const float* Wxh_b = (const float*)d_in[3];
  const float* Whh   = (const float*)d_in[4];
  const float* Whh_b = (const float*)d_in[5];
  const float* bhv   = (const float*)d_in[6];
  const float* fcw   = (const float*)d_in[7];
  const float* fcb   = (const float*)d_in[8];
  float* out = (float*)d_out;
  char*  ws  = (char*)d_ws;

  // workspace layout (256B-aligned chunks)
  const size_t OFF_XBF    = 0;                                  // 16384*512*2
  const size_t OFF_WXHBF  = OFF_XBF    + (size_t)SEQ * IN  * 2; // 1024*512*2
  const size_t OFF_WHHBF  = OFF_WXHBF  + (size_t)HID * IN  * 2; // 1024*1024*2
  const size_t OFF_FCBF   = OFF_WHHBF  + (size_t)HID * HID * 2; // 256*1024*2
  const size_t OFF_BSUM   = OFF_FCBF   + (size_t)OUT * HID * 2; // 1024*4
  const size_t OFF_HBUF   = OFF_BSUM   + (size_t)HID * 4;       // 2*1024*4
  const size_t OFF_CNT    = OFF_HBUF   + (size_t)2 * HID * 4;   // 256
  const size_t OFF_XH     = OFF_CNT    + 256;                   // 16384*1024*4
  const size_t OFF_OUTS   = OFF_XH     + (size_t)SEQ * HID * 4; // 16384*1024*2
  const size_t OFF_LOGITS = OFF_OUTS   + (size_t)SEQ * HID * 2; // 16384*256*4

  bf16_t*   Xbf    = (bf16_t*)(ws + OFF_XBF);
  bf16_t*   Wxhbf  = (bf16_t*)(ws + OFF_WXHBF);
  bf16_t*   Whhbf  = (bf16_t*)(ws + OFF_WHHBF);
  bf16_t*   Fcbf   = (bf16_t*)(ws + OFF_FCBF);
  float*    bsum   = (float*)  (ws + OFF_BSUM);
  float*    hbuf   = (float*)  (ws + OFF_HBUF);
  unsigned* cnt    = (unsigned*)(ws + OFF_CNT);
  float*    xh     = (float*)  (ws + OFF_XH);
  bf16_t*   outsbf = (bf16_t*) (ws + OFF_OUTS);
  float*    logits = (float*)  (ws + OFF_LOGITS);

  // 1) bf16 conversions + fused bias / h0 / barrier init
  cvt_f32_bf16<<<2048, 256, 0, stream>>>(X,   Xbf,   SEQ * IN);
  cvt_f32_bf16<<<512,  256, 0, stream>>>(Wxh, Wxhbf, HID * IN);
  cvt_f32_bf16<<<1024, 256, 0, stream>>>(Whh, Whhbf, HID * HID);
  cvt_f32_bf16<<<256,  256, 0, stream>>>(fcw, Fcbf,  OUT * HID);
  rnn_init<<<1, 1024, 0, stream>>>(Wxh_b, Whh_b, bhv, h0, bsum, hbuf, cnt);

  // 2) xh = X @ Wxh^T + bsum       (WMMA bf16)
  wmma_gemm_bias<<<dim3(HID / 64, SEQ / 128), 256, 0, stream>>>(
      Xbf, Wxhbf, bsum, xh, SEQ, HID, IN);

  // 3) sequential recurrence
  rnn_scan<<<SCAN_NB, 256, 0, stream>>>(
      xh, (const unsigned*)Whhbf, hbuf, outsbf, cnt, SEQ);

  // 4) logits = outs @ fc_w^T + fc_b   (WMMA bf16)
  wmma_gemm_bias<<<dim3(OUT / 64, SEQ / 128), 256, 0, stream>>>(
      outsbf, Fcbf, fcb, logits, SEQ, OUT, HID);

  // 5) row softmax
  softmax256<<<SEQ, 256, 0, stream>>>(logits, out);
}
